// PagedAttention_1855425872549
// MI455X (gfx1250) — matrix operand order repaired and checked by reading, not compiled, not run on hardware
//
#include <hip/hip_runtime.h>
#include <math.h>

// Problem constants (match reference setup_inputs()).
#define NUM_PAGES  1024
#define PAGE_SIZE  16
#define NUM_HEADS  32
#define HEAD_DIM   128
#define BATCH      32
#define MAX_BLOCKS 64
#define PART_STRIDE 130   // 128 acc + m + l  (floats per (b,h,split) partial)

typedef __attribute__((ext_vector_type(16))) _Float16 v16h;
typedef __attribute__((ext_vector_type(8)))  float    v8f;

// Build the 16x32 f16 A-fragment for one K page / one 32-dim chunk.
// ISA 7.12.2 (16-bit A 16x32): lane (m16, half) holds row M=m16 with
//   elems e in [0,8):  K = 8*half + e
//   elems e in [8,16): K = 16 + 8*half + (e-8)
// krow points at K[token=m16, head=h, dim=d0]; both runs are contiguous.
__device__ __forceinline__ v16h load_k_frag(const float* __restrict__ krow, int half) {
  v16h a;
  #pragma unroll
  for (int e = 0; e < 8; ++e) a[e]     = (_Float16)krow[8*half + e];
  #pragma unroll
  for (int e = 0; e < 8; ++e) a[8 + e] = (_Float16)krow[16 + 8*half + e];
  return a;
}

// One wave32 per workgroup: (split, head, batch). Flash-decoding partial over
// 128 tokens (8 pages) with online softmax; WMMA used as the GEMV engine.
__global__ __launch_bounds__(32)
void paged_attn_split_kernel(const float* __restrict__ q,    // [B,H,D]
                             const float* __restrict__ kc,   // [P,16,H,D]
                             const float* __restrict__ vc,   // [P,16,H,D]
                             const int*   __restrict__ bt,   // [B,MAX_BLOCKS]
                             const int*   __restrict__ cl,   // [B]
                             float* __restrict__ outp,       // partials or out
                             int splits)
{
  const int split = blockIdx.x;
  const int h     = blockIdx.y;
  const int b     = blockIdx.z;
  const int lane  = threadIdx.x;   // 0..31 (wave32)
  const int half  = lane >> 4;
  const int m16   = lane & 15;

  const int pagesPerSplit = MAX_BLOCKS / splits;
  const int pageBase      = split * pagesPerSplit;
  int ctx = cl[b]; if (ctx < 1) ctx = 1;

  // Query chunk broadcast fragments: B[k,n] = q[d0+k] * scale for all n.
  // B elem s of half h holds k = 8h+s (s<8) / 16+8h+(s-8); fold the 1/sqrt(D) scale into q.
  const float scale = 0.08838834764831845f; // 1/sqrt(128)
  const float* qh = q + ((size_t)b * NUM_HEADS + h) * HEAD_DIM;
  v16h qb[4];
  #pragma unroll
  for (int c = 0; c < 4; ++c) {
    #pragma unroll
    for (int n = 0; n < 16; ++n) qb[c][n] = (_Float16)(qh[c * 32 + 8 * half + (n & 7) + ((n >> 3) << 4)] * scale);
  }

  const v8f vzero = {0.f, 0.f, 0.f, 0.f, 0.f, 0.f, 0.f, 0.f};
  v8f acc[8];                         // 8 x (16x16 f32 C-frag) = 128 output dims
  #pragma unroll
  for (int c = 0; c < 8; ++c) acc[c] = vzero;
  float m_run = -1e30f, l_run = 0.f;

  const int  numPairs = pagesPerSplit >> 1;   // process 2 pages (32 tokens) per step
  const int* btb = bt + (size_t)b * MAX_BLOCKS + pageBase;

  for (int pair = 0; pair < numPairs; ++pair) {
    const int tokBase = (pageBase + 2 * pair) * PAGE_SIZE;  // global token idx
    if (tokBase >= ctx) break;                              // uniform early exit

    const int phys0 = btb[2 * pair + 0];
    const int phys1 = btb[2 * pair + 1];
    const float* k0 = kc + (((size_t)phys0 * PAGE_SIZE + m16) * NUM_HEADS + h) * HEAD_DIM;
    const float* k1 = kc + (((size_t)phys1 * PAGE_SIZE + m16) * NUM_HEADS + h) * HEAD_DIM;

    // Prefetch next pair's K row for this lane (global_prefetch_b8).
    if (pair + 1 < numPairs && tokBase + 32 < ctx) {
      const int pn = btb[2 * pair + 2];
      __builtin_prefetch(kc + (((size_t)pn * PAGE_SIZE + m16) * NUM_HEADS + h) * HEAD_DIM, 0, 1);
    }

    // ---- Scores: S[t] = q . K[t] (scaled), K-accumulated over D=128.
    v8f s0 = vzero, s1 = vzero;
    #pragma unroll
    for (int c = 0; c < 4; ++c) {
      v16h a0 = load_k_frag(k0 + c * 32, half);
      s0 = __builtin_amdgcn_wmma_f32_16x16x32_f16(false, a0, false, qb[c],
                                                  (short)0, s0, false, false);
    }
    #pragma unroll
    for (int c = 0; c < 4; ++c) {
      v16h a1 = load_k_frag(k1 + c * 32, half);
      s1 = __builtin_amdgcn_wmma_f32_16x16x32_f16(false, a1, false, qb[c],
                                                  (short)0, s1, false, false);
    }

    // C-frag: lane holds rows m = 8*half + r (all 16 cols identical).
    float e0[8], e1[8];
    float mx = -3e38f;
    #pragma unroll
    for (int r = 0; r < 8; ++r) {
      const int t0 = tokBase + 8 * half + r;       // page 2*pair
      const int t1 = t0 + PAGE_SIZE;               // page 2*pair+1
      e0[r] = (t0 < ctx) ? s0[r] : -3e38f;
      e1[r] = (t1 < ctx) ? s1[r] : -3e38f;
      mx = fmaxf(mx, fmaxf(e0[r], e1[r]));
    }
    mx = fmaxf(mx, __shfl_xor(mx, 16, 32));        // combine both half-lane groups
    const float m_new = fmaxf(m_run, mx);
    const float corr  = __expf(m_run - m_new);
    m_run = m_new;

    float p0[8], p1[8], lsum = 0.f;
    #pragma unroll
    for (int r = 0; r < 8; ++r) {
      p0[r] = __expf(e0[r] - m_new);               // masked -> exp(-3e38) == 0
      p1[r] = __expf(e1[r] - m_new);
      lsum += p0[r] + p1[r];
    }
    l_run = l_run * corr + (lsum + __shfl_xor(lsum, 16, 32));

    // ---- PV: A[m,k] = p[k] broadcast over rows m. Lane's own probs land on
    // exactly its A elements: e<8 -> page0 token 8*half+e, e>=8 -> page1.
    v16h pa;
    #pragma unroll
    for (int e = 0; e < 8; ++e) { pa[e] = (_Float16)p0[e]; pa[8 + e] = (_Float16)p1[e]; }

    // B col n = m16 (dim); elem e<8 -> page0 token 8*half+e, e>=8 -> page1 token 8*half+e-8.
    const float* v0 = vc + (((size_t)phys0 * PAGE_SIZE + 8 * half) * NUM_HEADS + h) * HEAD_DIM + m16;
    const float* v1 = vc + (((size_t)phys1 * PAGE_SIZE + 8 * half) * NUM_HEADS + h) * HEAD_DIM + m16;
    #pragma unroll
    for (int c2 = 0; c2 < 8; ++c2) {
      v16h vb;
      #pragma unroll
      for (int n = 0; n < 16; ++n) vb[n] = (_Float16)(n < 8 ? v0 : v1)[(size_t)(n & 7) * NUM_HEADS * HEAD_DIM + c2 * 16];
      acc[c2] = acc[c2] * corr;                    // online-softmax rescale
      acc[c2] = __builtin_amdgcn_wmma_f32_16x16x32_f16(false, pa, false, vb,
                                                       (short)0, acc[c2], false, false);
    }
  }

  // All C rows identical (A rows replicated): row 0 / lanes 0..15 carry result.
  if (splits == 1) {
    if (half == 0) {
      const float inv = 1.f / l_run;               // ctx >= 1 -> l_run > 0
      float* o = outp + ((size_t)b * NUM_HEADS + h) * HEAD_DIM;
      #pragma unroll
      for (int c2 = 0; c2 < 8; ++c2) o[c2 * 16 + m16] = acc[c2][0] * inv;
    }
  } else {
    float* p = outp + (((size_t)b * NUM_HEADS + h) * splits + split) * PART_STRIDE;
    if (half == 0) {
      #pragma unroll
      for (int c2 = 0; c2 < 8; ++c2) p[c2 * 16 + m16] = acc[c2][0];
    }
    if (lane == 0) { p[128] = m_run; p[129] = l_run; }
  }
}

// Merge split partials: out = sum_s exp(m_s-M)*acc_s / sum_s exp(m_s-M)*l_s.
__global__ __launch_bounds__(128)
void paged_attn_reduce_kernel(const float* __restrict__ part,
                              float* __restrict__ out, int splits)
{
  const int bh = blockIdx.x;        // b*H + h
  const int d  = threadIdx.x;       // 0..127
  const float* p = part + (size_t)bh * splits * PART_STRIDE;

  float M = -1e30f;
  for (int s = 0; s < splits; ++s) M = fmaxf(M, p[s * PART_STRIDE + 128]);
  float L = 0.f, val = 0.f;
  for (int s = 0; s < splits; ++s) {
    const float w = __expf(p[s * PART_STRIDE + 128] - M);  // empty split -> 0
    L   += w * p[s * PART_STRIDE + 129];
    val += w * p[s * PART_STRIDE + d];
  }
  out[(size_t)bh * HEAD_DIM + d] = val / L;
}

extern "C" void kernel_launch(void* const* d_in, const int* in_sizes, int n_in,
                              void* d_out, int out_size, void* d_ws, size_t ws_size,
                              hipStream_t stream) {
  const float* q  = (const float*)d_in[0];
  const float* kc = (const float*)d_in[1];
  const float* vc = (const float*)d_in[2];
  const int*   bt = (const int*)d_in[3];
  const int*   cl = (const int*)d_in[4];
  float* out = (float*)d_out;

  int splits = 8;  // 8192 waves; each split covers 128 tokens (8 pages)
  const size_t need = (size_t)BATCH * NUM_HEADS * splits * PART_STRIDE * sizeof(float);
  if (ws_size < need) splits = 1;  // deterministic fallback: direct-write path

  if (splits == 1) {
    dim3 grid(1, NUM_HEADS, BATCH);
    paged_attn_split_kernel<<<grid, 32, 0, stream>>>(q, kc, vc, bt, cl, out, 1);
  } else {
    dim3 grid(splits, NUM_HEADS, BATCH);
    paged_attn_split_kernel<<<grid, 32, 0, stream>>>(q, kc, vc, bt, cl,
                                                     (float*)d_ws, splits);
    paged_attn_reduce_kernel<<<BATCH * NUM_HEADS, 128, 0, stream>>>(
        (const float*)d_ws, out, splits);
  }
}
